// CrossAttention_38199439131420
// MI455X (gfx1250) — matrix operand order repaired and checked by reading, not compiled, not run on hardware
//
#include <hip/hip_runtime.h>

// CrossAttention on MI455X (gfx1250, wave32).
// Pipeline: [cvt f32->f16 x5] [GEMM Q] [GEMM KV (V stored transposed)]
//           [RoPE Q] [RoPE K] [flash-attn: async-to-LDS tiles + WMMA] [GEMM O].
// All matrix math on v_wmma_f32_16x16x32_f16 (f16 in, f32 accum).

typedef __attribute__((ext_vector_type(16))) _Float16 v16h;
typedef __attribute__((ext_vector_type(8)))  _Float16 v8h;
typedef __attribute__((ext_vector_type(8)))  float    v8f;

#define DEVINL __device__ __forceinline__

static DEVINL v8f wmma_f16(v16h a, v16h b, v8f c) {
  return __builtin_amdgcn_wmma_f32_16x16x32_f16(false, a, false, b, (short)0, c,
                                                false, false);
}

// ---------------- fragment loaders (ISA 7.12.2 layouts, wave32) ----------------
// A fragment 16x32 (MxK) f16: lanes 0-15 -> M=lane, K {0..7,16..23};
// lanes 16-31 -> M=lane-16, K {8..15,24..31}.  p0 = &X[row0*ld + k0].
static DEVINL v16h load_a_frag(const _Float16* p0, int ld, int lane) {
  const _Float16* p = p0 + (size_t)(lane & 15) * ld + ((lane >> 4) << 3);
  v8h lo = *reinterpret_cast<const v8h*>(p);        // K rel 0..7
  v8h hi = *reinterpret_cast<const v8h*>(p + 16);   // K rel 16..23
  v16h a;
#pragma unroll
  for (int e = 0; e < 8; ++e) { a[e] = lo[e]; a[8 + e] = hi[e]; }
  return a;
}

// B fragment 32x16 (KxN) from row-major [N,K] memory (B = M^T): lane holds
// column n = n0+(lane&15); K {0..7,16..23} (lanes 0-15) / {8..15,24..31} (lanes 16-31).
static DEVINL v16h load_b_nk(const _Float16* W, int ld, int n0, int k0, int lane) {
  const _Float16* p = W + (size_t)(n0 + (lane & 15)) * ld + k0 + ((lane >> 4) << 3);
  v8h lo = *reinterpret_cast<const v8h*>(p);
  v8h hi = *reinterpret_cast<const v8h*>(p + 16);
  v16h b;
#pragma unroll
  for (int e = 0; e < 8; ++e) { b[e] = lo[e]; b[8 + e] = hi[e]; }
  return b;
}

// ------------------------- f32 -> f16 convert pre-pass -------------------------
__global__ __launch_bounds__(256) void cvt_f16(const float* __restrict__ in,
                                               _Float16* __restrict__ out, int n) {
  int i = (blockIdx.x * blockDim.x + threadIdx.x) * 8;
  if (i >= n) return;
  const float4* p = reinterpret_cast<const float4*>(in + i);
  float4 a = p[0], b = p[1];
  v8h o;
  o[0] = (_Float16)a.x; o[1] = (_Float16)a.y; o[2] = (_Float16)a.z; o[3] = (_Float16)a.w;
  o[4] = (_Float16)b.x; o[5] = (_Float16)b.y; o[6] = (_Float16)b.z; o[7] = (_Float16)b.w;
  *reinterpret_cast<v8h*>(out + i) = o;
}

// ------------------------------- GEMM kernel ----------------------------------
// Out[M,N] = X[M,K] @ W[N,K]^T + bias.  256 thr = 8 waves; block tile 128x128;
// wave tile 64x32 (4x2 frags) -> 8 WMMA per 32-wide k-step.
template <typename TO>
__global__ __launch_bounds__(256) void gemm_wmma(const _Float16* __restrict__ X,
                                                 const _Float16* __restrict__ W,
                                                 const float* __restrict__ bias,
                                                 TO* __restrict__ O,
                                                 int M, int N, int K) {
  const int lane  = threadIdx.x & 31;
  const int wave  = threadIdx.x >> 5;
  const int mBase = blockIdx.x * 128 + (wave >> 2) * 64;
  const int nBase = blockIdx.y * 128 + (wave & 3) * 32;

  v8f acc[4][2];
#pragma unroll
  for (int mf = 0; mf < 4; ++mf)
#pragma unroll
    for (int nf = 0; nf < 2; ++nf)
      acc[mf][nf] = v8f{0.f, 0.f, 0.f, 0.f, 0.f, 0.f, 0.f, 0.f};

  for (int k0 = 0; k0 < K; k0 += 32) {
    v16h af[4], bf[2];
#pragma unroll
    for (int mf = 0; mf < 4; ++mf)
      af[mf] = load_a_frag(X + (size_t)(mBase + mf * 16) * K + k0, K, lane);
#pragma unroll
    for (int nf = 0; nf < 2; ++nf)
      bf[nf] = load_b_nk(W, K, nBase + nf * 16, k0, lane);
#pragma unroll
    for (int mf = 0; mf < 4; ++mf)
#pragma unroll
      for (int nf = 0; nf < 2; ++nf)
        acc[mf][nf] = wmma_f16(af[mf], bf[nf], acc[mf][nf]);
  }

#pragma unroll
  for (int mf = 0; mf < 4; ++mf)
#pragma unroll
    for (int nf = 0; nf < 2; ++nf) {
      const int n  = nBase + nf * 16 + (lane & 15);
      const float bv = bias[n];
      const int mb = mBase + mf * 16 + ((lane >> 4) << 3);
#pragma unroll
      for (int r = 0; r < 8; ++r)
        O[(size_t)(mb + r) * N + n] = (TO)(acc[mf][nf][r] + bv);
    }
}

// --------------------- KV projection: K normal, V transposed -------------------
// kv[m, n] for n<1024 -> kbuf[m, n]; for n>=1024 (V) -> vt[(b,h,d), t] so the
// attention PV B-operand becomes k-contiguous.  Transposed store is contiguous
// in t per lane (rows of a C fragment) -> single v8h store.
__global__ __launch_bounds__(256) void gemm_kv(const _Float16* __restrict__ X,
                                               const _Float16* __restrict__ W,
                                               const float* __restrict__ bias,
                                               _Float16* __restrict__ kbuf,
                                               _Float16* __restrict__ vt) {
  const int K = 1024;
  const int lane  = threadIdx.x & 31;
  const int wave  = threadIdx.x >> 5;
  const int mBase = blockIdx.x * 128 + (wave >> 2) * 64;
  const int nBase = blockIdx.y * 128 + (wave & 3) * 32;

  v8f acc[4][2];
#pragma unroll
  for (int mf = 0; mf < 4; ++mf)
#pragma unroll
    for (int nf = 0; nf < 2; ++nf)
      acc[mf][nf] = v8f{0.f, 0.f, 0.f, 0.f, 0.f, 0.f, 0.f, 0.f};

  for (int k0 = 0; k0 < K; k0 += 32) {
    v16h af[4], bf[2];
#pragma unroll
    for (int mf = 0; mf < 4; ++mf)
      af[mf] = load_a_frag(X + (size_t)(mBase + mf * 16) * K + k0, K, lane);
#pragma unroll
    for (int nf = 0; nf < 2; ++nf)
      bf[nf] = load_b_nk(W, K, nBase + nf * 16, k0, lane);
#pragma unroll
    for (int mf = 0; mf < 4; ++mf)
#pragma unroll
      for (int nf = 0; nf < 2; ++nf)
        acc[mf][nf] = wmma_f16(af[mf], bf[nf], acc[mf][nf]);
  }

#pragma unroll
  for (int mf = 0; mf < 4; ++mf)
#pragma unroll
    for (int nf = 0; nf < 2; ++nf) {
      const int n  = nBase + nf * 16 + (lane & 15);
      const float bv = bias[n];
      const int mb = mBase + mf * 16 + ((lane >> 4) << 3);
      if (n < 1024) {  // K half: row layout [8192, 1024] (uniform per fragment)
#pragma unroll
        for (int r = 0; r < 8; ++r)
          kbuf[(size_t)(mb + r) * 1024 + n] = (_Float16)(acc[mf][nf][r] + bv);
      } else {         // V half: transposed [(b*16+h)*64 + d][2048]
        const int hh = (n - 1024) >> 6, d = (n - 1024) & 63;
        const int bb = mb >> 11, t0 = mb & 2047;
        v8h vv;
#pragma unroll
        for (int r = 0; r < 8; ++r) vv[r] = (_Float16)(acc[mf][nf][r] + bv);
        *reinterpret_cast<v8h*>(
            vt + ((size_t)((bb * 16 + hh) * 64 + d)) * 2048 + t0) = vv;
      }
    }
}

// ------------------------------- RoPE kernel ----------------------------------
// out[j] = x[2j]*cos - x[2j+1]*sin ; out[j+32] = x[2j]*sin + x[2j+1]*cos.
__global__ __launch_bounds__(256) void rope_kernel(_Float16* __restrict__ buf,
                                                   const float* __restrict__ ct,
                                                   const float* __restrict__ st,
                                                   int rows, int ld) {
  int idx = blockIdx.x * blockDim.x + threadIdx.x;
  if (idx >= rows * 16) return;
  const int row = idx >> 4;
  const int hh  = idx & 15;
  const int pos = row & 2047;
  _Float16* p = buf + (size_t)row * ld + hh * 64;
  float x[64];
#pragma unroll
  for (int i = 0; i < 64; ++i) x[i] = (float)p[i];
#pragma unroll
  for (int j = 0; j < 32; ++j) {
    const float c = ct[pos * 32 + j];
    const float s = st[pos * 32 + j];
    p[j]      = (_Float16)(x[2 * j] * c - x[2 * j + 1] * s);
    p[32 + j] = (_Float16)(x[2 * j] * s + x[2 * j + 1] * c);
  }
}

// --------------------------- fused flash attention ----------------------------
// Block = 128 thr = 4 waves; one (b, h, 64 q-rows) per block.  K / V^T 64x64
// tiles are double-buffered in LDS via global_load_async_to_lds_b128 (ASYNCcnt),
// so every WMMA operand comes from registers or contiguous LDS b128 reads.
static DEVINL void async_tile(_Float16* lds, const _Float16* g, size_t gld,
                              int tid) {
  // 64x64 f16 = 8 KB = 512 x 16B chunks; 128 threads x 4 chunks.
#pragma unroll
  for (int j = 0; j < 4; ++j) {
    const int c = tid + j * 128;
    const int row = c >> 3, col = (c & 7) << 3;
    asm volatile("global_load_async_to_lds_b128 %0, %1, off"
                 :: "v"((unsigned)(uintptr_t)(lds + row * 64 + col)),
                    "v"(g + (size_t)row * gld + col)
                 : "memory");
  }
}

__global__ __launch_bounds__(128) void attn_fused(const _Float16* __restrict__ qb,
                                                  const _Float16* __restrict__ kb_,
                                                  const _Float16* __restrict__ vtb,
                                                  _Float16* __restrict__ ob) {
  const int Tq = 2048, Tkv = 2048, Dh = 64, D = 1024;
  const int tid  = threadIdx.x;
  const int lane = tid & 31;
  const int wave = tid >> 5;
  const int qt = blockIdx.x & 31;
  const int h  = (blockIdx.x >> 5) & 15;
  const int b  = blockIdx.x >> 9;
  const int qRow = b * Tq + qt * 64 + wave * 16;

  const _Float16* qp     = qb  + (size_t)qRow * D + h * Dh;
  const _Float16* kbase  = kb_ + (size_t)(b * Tkv) * D + h * Dh;          // [t, d]
  const _Float16* vtbase = vtb + (size_t)((b * 16 + h) * 64) * 2048;      // [d, t]

  __shared__ __align__(16) _Float16 kT[2][64 * 64];
  __shared__ __align__(16) _Float16 vT[2][64 * 64];
  __shared__ __align__(16) _Float16 pT[4][16 * 64];
  _Float16* pw = pT[wave];

  // resident Q fragments (16 rows x 64)
  v16h qa[2];
  qa[0] = load_a_frag(qp, D, lane);
  qa[1] = load_a_frag(qp + 32, D, lane);

  float mrow[8], lrow[8];
  v8f oacc[4];
#pragma unroll
  for (int r = 0; r < 8; ++r) { mrow[r] = -1e30f; lrow[r] = 0.f; }
#pragma unroll
  for (int df = 0; df < 4; ++df)
    oacc[df] = v8f{0.f, 0.f, 0.f, 0.f, 0.f, 0.f, 0.f, 0.f};

  const float scale = 0.125f;  // 1/sqrt(64)

  // prefetch tile 0
  async_tile(kT[0], kbase, (size_t)D, tid);
  async_tile(vT[0], vtbase, (size_t)2048, tid);

  for (int it = 0; it < Tkv / 64; ++it) {
    const int cur = it & 1;
    if (it < Tkv / 64 - 1) {  // stage next tile, then wait for current only
      async_tile(kT[cur ^ 1], kbase + (size_t)(it + 1) * 64 * D, (size_t)D, tid);
      async_tile(vT[cur ^ 1], vtbase + (it + 1) * 64, (size_t)2048, tid);
      asm volatile("s_wait_asynccnt 0x8" ::: "memory");
    } else {
      asm volatile("s_wait_asynccnt 0x0" ::: "memory");
    }
    __syncthreads();

    // ---- S = Q * K^T  (16 x 64), B operand from LDS (kT is [t, d]) ----
    v8f s[4];
#pragma unroll
    for (int nf = 0; nf < 4; ++nf) {
      s[nf] = v8f{0.f, 0.f, 0.f, 0.f, 0.f, 0.f, 0.f, 0.f};
#pragma unroll
      for (int ks = 0; ks < 2; ++ks)
        s[nf] = wmma_f16(qa[ks], load_b_nk(kT[cur], 64, nf * 16, ks * 32, lane),
                         s[nf]);
    }

    // ---- online softmax; row m = r + 8*(lane>>4), cols across 16 lanes ----
    float rmax[8];
#pragma unroll
    for (int r = 0; r < 8; ++r)
      rmax[r] = fmaxf(fmaxf(s[0][r], s[1][r]), fmaxf(s[2][r], s[3][r])) * scale;
#pragma unroll
    for (int m = 1; m <= 8; m <<= 1)
#pragma unroll
      for (int r = 0; r < 8; ++r)
        rmax[r] = fmaxf(rmax[r], __shfl_xor(rmax[r], m, 32));

    float corr[8], rs[8];
#pragma unroll
    for (int r = 0; r < 8; ++r) {
      const float nm = fmaxf(mrow[r], rmax[r]);
      corr[r] = __expf(mrow[r] - nm);
      mrow[r] = nm;
      rs[r]   = 0.f;
    }

    // P = exp(S*scale - m) -> per-wave LDS (C-layout -> row-major)
#pragma unroll
    for (int nf = 0; nf < 4; ++nf)
#pragma unroll
      for (int r = 0; r < 8; ++r) {
        const float pv = __expf(s[nf][r] * scale - mrow[r]);
        rs[r] += pv;
        pw[(size_t)(r + ((lane >> 4) << 3)) * 64 + nf * 16 + (lane & 15)] =
            (_Float16)pv;
      }
#pragma unroll
    for (int m = 1; m <= 8; m <<= 1)
#pragma unroll
      for (int r = 0; r < 8; ++r) rs[r] += __shfl_xor(rs[r], m, 32);
#pragma unroll
    for (int r = 0; r < 8; ++r) lrow[r] = lrow[r] * corr[r] + rs[r];
#pragma unroll
    for (int df = 0; df < 4; ++df)
#pragma unroll
      for (int r = 0; r < 8; ++r) oacc[df][r] *= corr[r];

    // ---- O += P * V ; B operand from LDS (vT is [d, t] = [N, K]) ----
#pragma unroll
    for (int ks = 0; ks < 2; ++ks) {
      const v16h pa = load_a_frag(pw + ks * 32, 64, lane);
#pragma unroll
      for (int df = 0; df < 4; ++df)
        oacc[df] = wmma_f16(pa, load_b_nk(vT[cur], 64, df * 16, ks * 32, lane),
                            oacc[df]);
    }
    __syncthreads();  // protect buffer cur before it is re-staged next iter
  }

  _Float16* orow = ob + (size_t)qRow * D + h * Dh;
#pragma unroll
  for (int df = 0; df < 4; ++df)
#pragma unroll
    for (int r = 0; r < 8; ++r) {
      const int m = ((lane >> 4) << 3) + r;
      const int n = df * 16 + (lane & 15);
      orow[(size_t)m * D + n] = (_Float16)(oacc[df][r] / lrow[r]);
    }
}

// --------------------------------- launcher -----------------------------------
extern "C" void kernel_launch(void* const* d_in, const int* in_sizes, int n_in,
                              void* d_out, int out_size, void* d_ws, size_t ws_size,
                              hipStream_t stream) {
  const float* x    = (const float*)d_in[0];
  const float* ctx  = (const float*)d_in[1];
  const float* Wq   = (const float*)d_in[2];
  const float* bq   = (const float*)d_in[3];
  const float* Wkv  = (const float*)d_in[4];
  const float* bkv  = (const float*)d_in[5];
  const float* Wo   = (const float*)d_in[6];
  const float* bo   = (const float*)d_in[7];
  const float* ctab = (const float*)d_in[8];
  const float* stab = (const float*)d_in[9];
  float* out = (float*)d_out;

  const int BT = 4 * 2048;  // 8192 token rows
  _Float16* p = (_Float16*)d_ws;
  _Float16* xh   = p; p += (size_t)BT * 1024;        // x in f16
  _Float16* ch   = p; p += (size_t)BT * 1024;        // context in f16
  _Float16* wqh  = p; p += (size_t)1024 * 1024;
  _Float16* wkvh = p; p += (size_t)2048 * 1024;
  _Float16* woh  = p; p += (size_t)1024 * 1024;
  _Float16* qbuf = p; p += (size_t)BT * 1024;        // Q  [8192,1024]
  _Float16* kbuf = p; p += (size_t)BT * 1024;        // K  [8192,1024]
  _Float16* vtbf = p; p += (size_t)BT * 1024;        // V^T [(b,h,d)=4096+..][2048]
  _Float16* obuf = p;                                 // attn out [8192,1024]

  // one-time f32 -> f16 down-converts (removes cvts from GEMM hot loops)
  cvt_f16<<<(BT * 1024) / 8 / 256, 256, 0, stream>>>(x,   xh,   BT * 1024);
  cvt_f16<<<(BT * 1024) / 8 / 256, 256, 0, stream>>>(ctx, ch,   BT * 1024);
  cvt_f16<<<(1024 * 1024) / 8 / 256, 256, 0, stream>>>(Wq,  wqh,  1024 * 1024);
  cvt_f16<<<(2048 * 1024) / 8 / 256, 256, 0, stream>>>(Wkv, wkvh, 2048 * 1024);
  cvt_f16<<<(1024 * 1024) / 8 / 256, 256, 0, stream>>>(Wo,  woh,  1024 * 1024);

  // Q projection
  gemm_wmma<_Float16><<<dim3(BT / 128, 1024 / 128), 256, 0, stream>>>(
      xh, wqh, bq, qbuf, BT, 1024, 1024);
  // KV projection (K row-major, V transposed)
  gemm_kv<<<dim3(BT / 128, 2048 / 128), 256, 0, stream>>>(ch, wkvh, bkv, kbuf,
                                                          vtbf);
  // RoPE on Q and K
  rope_kernel<<<(BT * 16) / 256, 256, 0, stream>>>(qbuf, ctab, stab, BT, 1024);
  rope_kernel<<<(BT * 16) / 256, 256, 0, stream>>>(kbuf, ctab, stab, BT, 1024);
  // fused attention
  attn_fused<<<4 * 16 * (2048 / 64), 128, 0, stream>>>(qbuf, kbuf, vtbf, obuf);
  // output projection (f32 result)
  gemm_wmma<float><<<dim3(BT / 128, 1024 / 128), 256, 0, stream>>>(
      obuf, woh, bo, out, BT, 1024, 1024);
}
